// SymbolicPlanner_4131758539380
// MI455X (gfx1250) — matrix operand order repaired and checked by reading, not compiled, not run on hardware
//
#include <hip/hip_runtime.h>
#include <hip/hip_bf16.h>
#include <stdint.h>

#define BB 2048
#define ZZ 128
#define AA 4
#define DD 50000
#define NSEG 8
#define SEGLEN 6272   // 32-aligned so async b128 global addresses stay 16B aligned

typedef __attribute__((ext_vector_type(16))) __bf16 v16bf;
typedef __attribute__((ext_vector_type(8)))  float  v8f;

union Frag { v16bf v; uint4 u[2]; };

static __device__ __forceinline__ unsigned short f2bf(float f) {
  unsigned u = __float_as_uint(f);
  u += 0x7fffu + ((u >> 16) & 1u);   // RNE
  return (unsigned short)(u >> 16);
}

static __device__ __forceinline__ void async_b128(unsigned lds, unsigned long long g) {
  // GLOBAL_LOAD_ASYNC_TO_LDS_B128: VDST = per-lane LDS byte address, VADDR = 64b global addr
  asm volatile("global_load_async_to_lds_b128 %0, %1, off" :: "v"(lds), "v"(g) : "memory");
}

// ---------------- prep: row-normalize f32 -> bf16 (used for q and keys) ------
__global__ void rownorm_bf16_kernel(const float* __restrict__ src,
                                    unsigned short* __restrict__ dst) {
  int row  = blockIdx.x * 8 + (threadIdx.x >> 5);
  int lane = threadIdx.x & 31;
  const float4 f = ((const float4*)(src + (size_t)row * ZZ))[lane];
  float ss = f.x*f.x + f.y*f.y + f.z*f.z + f.w*f.w;
  #pragma unroll
  for (int off = 1; off < 32; off <<= 1) ss += __shfl_xor(ss, off, 32);
  float inv = 1.0f / (sqrtf(ss) + 1e-8f);
  ushort4 o;
  o.x = f2bf(f.x*inv); o.y = f2bf(f.y*inv); o.z = f2bf(f.z*inv); o.w = f2bf(f.w*inv);
  ((ushort4*)(dst + (size_t)row * ZZ))[lane] = o;
}

// ---------------- prep: vals [A][D][Z] f32 -> vT [A][Z][D] bf16 --------------
__global__ void vT_bf16_kernel(const float* __restrict__ vals,
                               unsigned short* __restrict__ vt) {
  __shared__ float tile[32][33];
  int a  = blockIdx.z;
  int z0 = blockIdx.y * 32;
  int d0 = blockIdx.x * 32;
  int t  = threadIdx.x;
  int dl = t >> 3, zq = t & 7;
  if (d0 + dl < DD) {
    const float4 f = *(const float4*)(vals + ((size_t)a*DD + d0 + dl)*ZZ + z0 + zq*4);
    tile[dl][zq*4+0] = f.x; tile[dl][zq*4+1] = f.y;
    tile[dl][zq*4+2] = f.z; tile[dl][zq*4+3] = f.w;
  }
  __syncthreads();
  int zl = t >> 3, dq = t & 7;
  unsigned short* out = vt + ((size_t)a*ZZ + z0 + zl)*DD + d0;
  if (d0 + dq*4 + 3 < DD) {
    ushort4 o;
    o.x = f2bf(tile[dq*4+0][zl]); o.y = f2bf(tile[dq*4+1][zl]);
    o.z = f2bf(tile[dq*4+2][zl]); o.w = f2bf(tile[dq*4+3][zl]);
    *(ushort4*)(out + dq*4) = o;
  } else {
    for (int j = 0; j < 4; ++j)
      if (d0 + dq*4 + j < DD) out[dq*4+j] = f2bf(tile[dq*4+j][zl]);
  }
}

// ---------------- deterministic per-action compaction (1 wave per action) ----
__global__ void compact_kernel(const int* __restrict__ ba,
                               int* __restrict__ idx, int* __restrict__ counts) {
  int a = threadIdx.x >> 5, lane = threadIdx.x & 31;
  int cnt = 0;
  for (int b0 = 0; b0 < BB; b0 += 32) {
    bool mt = (ba[b0 + lane] == a);
    unsigned mask = (unsigned)__ballot(mt);
    if (mt) idx[a*BB + cnt + __popc(mask & ((1u << lane) - 1u))] = b0 + lane;
    cnt += __popc(mask);
  }
  if (lane == 0) counts[a] = cnt;
}

// ---------------- main flash-attention kernel --------------------------------
__global__ __launch_bounds__(256) void flash_kernel(
    const unsigned short* __restrict__ qb,
    const unsigned short* __restrict__ knb,
    const unsigned short* __restrict__ vtb,
    const int* __restrict__ idx, const int* __restrict__ counts,
    float* __restrict__ pm, float* __restrict__ pl, float* __restrict__ pacc) {
  __shared__ unsigned short kbuf[2][32*ZZ];   // 32 keys x 128 dims, row-major
  __shared__ unsigned short vbuf[2][ZZ*32];   // 128 z x 32 keys (pre-transposed V)
  __shared__ unsigned short pbuf[8][16*32];   // per-wave P staging (16 rows x 32 keys)

  const int a = blockIdx.z, seg = blockIdx.y, tilei = blockIdx.x;
  const int n = counts[a];
  const int base = tilei * 128;
  if (base >= n) return;                       // uniform exit, before any barrier

  const int t = threadIdx.x;
  const int wave = t >> 5, lane = t & 31, lh = lane & 15, half = lane >> 4;
  const int wbase = base + wave * 16;

  const int segbase = seg * SEGLEN;
  int limit = segbase + SEGLEN; if (limit > DD) limit = DD;
  const int nch = (limit - segbase + 31) >> 5;

  int gpos = wbase + lh; if (gpos >= n) gpos = n - 1;   // pad with a valid row
  const int ridx = idx[a*BB + gpos];

  // Q A-fragments (16x32 bf16 per K-tile): lane holds row M=lh, two 16B runs
  Frag qa[4];
  {
    const uint4* qr = (const uint4*)(qb + (size_t)ridx * ZZ);
    #pragma unroll
    for (int kt = 0; kt < 4; ++kt) {
      qa[kt].u[0] = qr[kt*4 + half];
      qa[kt].u[1] = qr[kt*4 + 2 + half];
    }
  }

  float m[8], l[8];
  v8f acc[8];
  #pragma unroll
  for (int i = 0; i < 8; ++i) { m[i] = -3.0e38f; l[i] = 0.0f; }
  #pragma unroll
  for (int nn2 = 0; nn2 < 8; ++nn2) {
    #pragma unroll
    for (int i = 0; i < 8; ++i) acc[nn2][i] = 0.0f;
  }

  const char* knA = (const char*)knb + (size_t)a * DD * ZZ * 2;
  const char* vtA = (const char*)vtb + (size_t)a * ZZ * DD * 2;

  auto issue = [&](int buf, int kb0) {
    int r = t >> 3, q8 = t & 7;   // K chunk: 32 rows x 256B
    unsigned long long gk = (unsigned long long)(uintptr_t)(knA + (size_t)(kb0 + r)*256 + q8*32);
    unsigned lk = (unsigned)(uintptr_t)(&kbuf[buf][0]) + (unsigned)(r*256 + q8*32);
    async_b128(lk, gk); async_b128(lk + 16, gk + 16);
    int z = t >> 1, hh = t & 1;   // V chunk: 128 z-rows x 64B strided slices of vT
    unsigned long long gv = (unsigned long long)(uintptr_t)(vtA + (size_t)z*(DD*2) + (size_t)(kb0 + hh*16)*2);
    unsigned lv = (unsigned)(uintptr_t)(&vbuf[buf][0]) + (unsigned)(z*64 + hh*32);
    async_b128(lv, gv); async_b128(lv + 16, gv + 16);
  };

  issue(0, segbase);
  for (int c = 0; c < nch; ++c) {
    const int buf = c & 1;
    const bool more = (c + 1) < nch;
    if (more) issue(buf ^ 1, segbase + (c + 1)*32);
    if (more) asm volatile("s_wait_asynccnt 0x4" ::: "memory");   // current chunk landed
    else      asm volatile("s_wait_asynccnt 0x0" ::: "memory");
    __syncthreads();

    const int keybase = segbase + c*32;

    // ---- sims: S(16x32) = Q(16x128) @ K^T(128x32), bf16 WMMA f32 acc ----
    v8f s0, s1;
    #pragma unroll
    for (int i = 0; i < 8; ++i) { s0[i] = 0.0f; s1[i] = 0.0f; }
    const uint4* kb4 = (const uint4*)&kbuf[buf][0];
    #pragma unroll
    for (int kt = 0; kt < 4; ++kt) {
      Frag b0, b1;
      b0.u[0] = kb4[lh*16 + kt*4 + half];
      b0.u[1] = kb4[lh*16 + kt*4 + 2 + half];
      s0 = __builtin_amdgcn_wmma_f32_16x16x32_bf16(false, qa[kt].v, false, b0.v,
                                                   (short)0, s0, false, false);
      b1.u[0] = kb4[(16+lh)*16 + kt*4 + half];
      b1.u[1] = kb4[(16+lh)*16 + kt*4 + 2 + half];
      s1 = __builtin_amdgcn_wmma_f32_16x16x32_bf16(false, qa[kt].v, false, b1.v,
                                                   (short)0, s1, false, false);
    }
    // mask columns past the segment end (lane's column = keybase(+16)+lh)
    if (keybase + lh >= limit) {
      #pragma unroll
      for (int i = 0; i < 8; ++i) s0[i] = -1.0e30f;
    }
    if (keybase + 16 + lh >= limit) {
      #pragma unroll
      for (int i = 0; i < 8; ++i) s1[i] = -1.0e30f;
    }

    // ---- online softmax (row = element i + 8*half, reduce across 16 lanes) ----
    float al[8];
    #pragma unroll
    for (int i = 0; i < 8; ++i) {
      float v = fmaxf(s0[i], s1[i]);
      #pragma unroll
      for (int off = 1; off < 16; off <<= 1) v = fmaxf(v, __shfl_xor(v, off, 16));
      float mn = fmaxf(m[i], v);
      al[i] = __expf(m[i] - mn);
      m[i] = mn;
    }
    #pragma unroll
    for (int i = 0; i < 8; ++i) {
      float e0 = __expf(s0[i] - m[i]);
      float e1 = __expf(s1[i] - m[i]);
      s0[i] = e0; s1[i] = e1;
      float rs = e0 + e1;
      #pragma unroll
      for (int off = 1; off < 16; off <<= 1) rs += __shfl_xor(rs, off, 16);
      l[i] = l[i]*al[i] + rs;
    }

    // ---- transpose P from C-layout to A-layout through LDS ----
    unsigned short* pb = &pbuf[wave][0];
    #pragma unroll
    for (int i = 0; i < 8; ++i) {
      int r = i + half*8;
      pb[r*32 + lh]      = f2bf(s0[i]);
      pb[r*32 + 16 + lh] = f2bf(s1[i]);
    }
    asm volatile("s_wait_dscnt 0x0" ::: "memory");   // wave-internal LDS fence
    Frag pa;
    {
      const uint4* pb4 = (const uint4*)&pbuf[wave][0];
      pa.u[0] = pb4[lh*4 + half];
      pa.u[1] = pb4[lh*4 + 2 + half];
    }

    // ---- acc = acc*alpha + P(16x32) @ V(32x128) ----
    #pragma unroll
    for (int nn2 = 0; nn2 < 8; ++nn2) {
      #pragma unroll
      for (int i = 0; i < 8; ++i) acc[nn2][i] *= al[i];
    }

    const uint4* vb4 = (const uint4*)&vbuf[buf][0];
    #pragma unroll
    for (int nn2 = 0; nn2 < 8; ++nn2) {
      Frag vb;
      int z = nn2*16 + lh;
      vb.u[0] = vb4[z*4 + half];
      vb.u[1] = vb4[z*4 + 2 + half];
      acc[nn2] = __builtin_amdgcn_wmma_f32_16x16x32_bf16(false, pa.v, false, vb.v,
                                                         (short)0, acc[nn2], false, false);
    }
    __syncthreads();
  }

  // ---- write segment partials ----
  #pragma unroll
  for (int i = 0; i < 8; ++i) {
    int M = i + half*8;
    int gp = wbase + M;
    if (gp < n) {
      int b = idx[a*BB + gp];
      size_t po = ((size_t)b*NSEG + seg)*ZZ;
      #pragma unroll
      for (int nn2 = 0; nn2 < 8; ++nn2)
        pacc[po + nn2*16 + lh] = acc[nn2][i];
      if (lh == 0) { pm[b*NSEG + seg] = m[i]; pl[b*NSEG + seg] = l[i]; }
    }
  }
}

// ---------------- merge segment partials, per-sample squared error -----------
__global__ void combine_kernel(const float* __restrict__ pm, const float* __restrict__ pl,
                               const float* __restrict__ pacc, const float* __restrict__ bnext,
                               float* __restrict__ sqerr) {
  int b = blockIdx.x*8 + (threadIdx.x >> 5);
  int lane = threadIdx.x & 31;
  float mm = -3.0e38f;
  #pragma unroll
  for (int s = 0; s < NSEG; ++s) mm = fmaxf(mm, pm[b*NSEG + s]);
  float w[NSEG]; float Ls = 0.0f;
  #pragma unroll
  for (int s = 0; s < NSEG; ++s) { w[s] = __expf(pm[b*NSEG + s] - mm); Ls += pl[b*NSEG + s]*w[s]; }
  float inv = 1.0f / Ls;
  float px = 0, py = 0, pz = 0, pw = 0;
  #pragma unroll
  for (int s = 0; s < NSEG; ++s) {
    const float4 v = ((const float4*)(pacc + ((size_t)b*NSEG + s)*ZZ))[lane];
    px += v.x*w[s]; py += v.y*w[s]; pz += v.z*w[s]; pw += v.w*w[s];
  }
  const float4 nx = ((const float4*)(bnext + (size_t)b*ZZ))[lane];
  float dx = px*inv - nx.x, dy = py*inv - nx.y, dz = pz*inv - nx.z, dw = pw*inv - nx.w;
  float sq = dx*dx + dy*dy + dz*dz + dw*dw;
  #pragma unroll
  for (int off = 1; off < 32; off <<= 1) sq += __shfl_xor(sq, off, 32);
  if (lane == 0) sqerr[b] = sq;
}

// ---------------- fixed-order final reduction --------------------------------
__global__ void reduce_kernel(const float* __restrict__ sqerr, float* __restrict__ out) {
  __shared__ float sh[256];
  int t = threadIdx.x;
  float s = 0.0f;
  for (int i = 0; i < BB/256; ++i) s += sqerr[t + 256*i];
  sh[t] = s; __syncthreads();
  for (int st = 128; st > 0; st >>= 1) {
    if (t < st) sh[t] += sh[t + st];
    __syncthreads();
  }
  if (t == 0) out[0] = sh[0] * (1.0f / ((float)BB * (float)ZZ));
}

extern "C" void kernel_launch(void* const* d_in, const int* in_sizes, int n_in,
                              void* d_out, int out_size, void* d_ws, size_t ws_size,
                              hipStream_t stream) {
  (void)in_sizes; (void)n_in; (void)out_size; (void)ws_size;
  const float* bx   = (const float*)d_in[0];
  const float* bn   = (const float*)d_in[1];
  const float* keys = (const float*)d_in[2];
  const float* vals = (const float*)d_in[3];
  const int*   ba   = (const int*)d_in[4];

  char* ws = (char*)d_ws;
  size_t off = 0;
  auto alloc = [&](size_t bytes) { size_t o = off; off = (off + bytes + 255) & ~(size_t)255; return o; };
  unsigned short* qb   = (unsigned short*)(ws + alloc((size_t)BB*ZZ*2));
  unsigned short* knb  = (unsigned short*)(ws + alloc((size_t)AA*DD*ZZ*2));
  unsigned short* vtb  = (unsigned short*)(ws + alloc((size_t)AA*ZZ*DD*2));
  float* pacc          = (float*)(ws + alloc((size_t)BB*NSEG*ZZ*4));   // directly after vtb (tail overreads land here)
  float* pm            = (float*)(ws + alloc((size_t)BB*NSEG*4));
  float* pl            = (float*)(ws + alloc((size_t)BB*NSEG*4));
  int*   idx           = (int*)(ws + alloc((size_t)AA*BB*4));
  int*   counts        = (int*)(ws + alloc(256));
  float* sq            = (float*)(ws + alloc((size_t)BB*4));

  rownorm_bf16_kernel<<<BB/8, 256, 0, stream>>>(bx, qb);
  rownorm_bf16_kernel<<<(AA*DD)/8, 256, 0, stream>>>(keys, knb);
  vT_bf16_kernel<<<dim3((DD + 31)/32, ZZ/32, AA), 256, 0, stream>>>(vals, vtb);
  compact_kernel<<<1, 128, 0, stream>>>(ba, idx, counts);
  flash_kernel<<<dim3(BB/128, NSEG, AA), 256, 0, stream>>>(qb, knb, vtb, idx, counts, pm, pl, pacc);
  combine_kernel<<<BB/8, 256, 0, stream>>>(pm, pl, pacc, bn, sq);
  reduce_kernel<<<1, 256, 0, stream>>>(sq, (float*)d_out);
}
